// OfficialPSMNET_13580686590298
// MI455X (gfx1250) — compile-verified, matrix operand order checked by reading, not run on hardware
//
#include <hip/hip_runtime.h>

// ---------------------------------------------------------------------------
// Types for CDNA5 WMMA
// ---------------------------------------------------------------------------
typedef __attribute__((ext_vector_type(16))) __bf16 v16bf;
typedef __attribute__((ext_vector_type(8)))  float  v8f;

union FragBF {
    v16bf v;
    unsigned int u32[8];
};

__device__ __forceinline__ unsigned short f2bf(float f) {
    unsigned int u = __float_as_uint(f);
    unsigned int r = u + 0x7FFFu + ((u >> 16) & 1u);   // round-to-nearest-even
    return (unsigned short)(r >> 16);
}
__device__ __forceinline__ float bf2f(unsigned short u) {
    return __uint_as_float(((unsigned int)u) << 16);
}

// ---------------------------------------------------------------------------
// uint4 zero-fill (activation halo region init)
// ---------------------------------------------------------------------------
__global__ void __launch_bounds__(256) k_zero(uint4* __restrict__ p, unsigned int n16) {
    unsigned int i = blockIdx.x * 256 + threadIdx.x;
    if (i < n16) {
        uint4 z; z.x = 0u; z.y = 0u; z.z = 0u; z.w = 0u;
        p[i] = z;
    }
}

// ---------------------------------------------------------------------------
// Weight repack: f32 (mode0: [Cout,Cin,27] | mode1 convT: [Cin,Cout,27], tap
// flipped) -> bf16 canonical [Coutp][27][Cin], zero-padded to Coutp.
// ---------------------------------------------------------------------------
__global__ void __launch_bounds__(256) k_repack(const float* __restrict__ w,
                                                unsigned short* __restrict__ o,
                                                int Cin, int Cout, int mode, int n) {
    int i = blockIdx.x * 256 + threadIdx.x;
    if (i >= n) return;                        // n = Coutp*27*Cin
    int ci  = i % Cin;  int t = i / Cin;
    int tap = t % 27;   int co = t / 27;
    float v = 0.f;
    if (co < Cout) {
        if (mode == 0) {
            v = w[((size_t)co * Cin + ci) * 27 + tap];
        } else {
            int kd = tap / 9, kh = (tap / 3) % 3, kw = tap % 3;
            int ft = (2 - kd) * 9 + (2 - kh) * 3 + (2 - kw);
            v = w[((size_t)ci * Cout + co) * 27 + ft];
        }
    }
    o[i] = f2bf(v);
}

// ---------------------------------------------------------------------------
// Cost volume: conv_out f32 (2,32,64,128) -> cost bf16 channel-last PADDED
// [50,66,130,64] (interior write; halo pre-zeroed)
// ---------------------------------------------------------------------------
__global__ void __launch_bounds__(256) k_cost(const float* __restrict__ co,
                                              unsigned short* __restrict__ cost,
                                              int n) {
    int i = blockIdx.x * 256 + threadIdx.x;
    if (i >= n) return;                       // n = 64 * 48*64*128
    int c   = i & 63;
    int vox = i >> 6;
    int w = vox & 127;
    int h = (vox >> 7) & 63;
    int d = vox >> 13;
    float v = 0.f;
    if (w >= d) {
        int ws = (c < 32) ? w : (w - d);
        v = co[((size_t)c * 64 + h) * 128 + ws];
    }
    int pidx = ((((d + 1) * 66 + (h + 1)) * 130 + (w + 1)) << 6) + c;
    cost[pidx] = f2bf(v);
}

// ---------------------------------------------------------------------------
// Implicit-GEMM conv3d / convT3d on PADDED channel-last bf16 input.
// One wave32 computes a 16(voxels along W) x 16(Cout) tile with
// v_wmma_f32_16x16x32_bf16.
//   mode 0: regular conv (pad baked into halo -> no bounds checks at all)
//   mode 1: transposed conv (lhs_dilation 2): uniform d/h tap skip + per-lane
//           zero-buffer address select for the w parity mask
//   outLayout 0: channel-last f32 [vox*Cout + co]   (coalesced, unpadded)
//   outLayout 1: channel-major f32 [co*V + vox]     (final head -> d_out)
// ---------------------------------------------------------------------------
__global__ void __launch_bounds__(256) k_conv(
    const unsigned short* __restrict__ in,     // bf16 [Dp,Hp,Wp,Cin] padded
    const unsigned short* __restrict__ wgt,    // bf16 [Coutp][27][Cin]
    const unsigned short* __restrict__ zbuf,   // >=256B of zeros
    float* __restrict__ out,
    int Cin, int Hp, int Wp,                   // padded input H+2, W+2
    int Di, int Hi, int Wi,                    // unpadded input dims (mode 1)
    int Cout, int Do, int Ho, int Wo,
    int stride, int mode, int outLayout, int acc, int nTiles)
{
    int lane = threadIdx.x & 31;
    int wave = threadIdx.x >> 5;
    int tile = blockIdx.x * 8 + wave;
    if (tile >= nTiles) return;                // uniform per wave: EXEC stays full

    int nTw = Wo >> 4;
    int tw  = tile % nTw; tile /= nTw;
    int ho  = tile % Ho;  tile /= Ho;
    int dd  = tile % Do;  tile /= Do;
    int tco = tile;

    int mrow  = lane & 15;          // A row (voxel) / B col (cout)
    int khalf = lane >> 4;          // which K half this lane holds
    int wo    = tw * 16 + mrow;     // output w for the A row
    int co    = tco * 16 + mrow;    // output channel for the B col

    v8f c = {0.f, 0.f, 0.f, 0.f, 0.f, 0.f, 0.f, 0.f};

    const int rowC   = Wp * Cin;
    const int planeC = Hp * rowC;
    const unsigned short* wbase = wgt + (size_t)co * 27 * Cin + khalf * 8;

    if (mode == 0) {
        for (int kd = 0; kd < 3; ++kd)
        for (int kh = 0; kh < 3; ++kh)
        for (int kw = 0; kw < 3; ++kw) {
            int di = dd * stride + kd;          // halo-padded: always in bounds
            int hi = ho * stride + kh;
            int wi = wo * stride + kw;
            int tap = kd * 9 + kh * 3 + kw;
            const unsigned short* ap = in + (size_t)(di * planeC + hi * rowC + wi * Cin)
                                          + khalf * 8;
            const unsigned short* wp = wbase + tap * Cin;
            __builtin_prefetch(ap, 0, 1);
            for (int c0 = 0; c0 < Cin; c0 += 32) {
                uint4 a0 = *(const uint4*)(ap + c0);
                uint4 a1 = *(const uint4*)(ap + c0 + 16);
                uint4 b0 = *(const uint4*)(wp + c0);
                uint4 b1 = *(const uint4*)(wp + c0 + 16);
                FragBF a, b;
                a.u32[0] = a0.x; a.u32[1] = a0.y; a.u32[2] = a0.z; a.u32[3] = a0.w;
                a.u32[4] = a1.x; a.u32[5] = a1.y; a.u32[6] = a1.z; a.u32[7] = a1.w;
                b.u32[0] = b0.x; b.u32[1] = b0.y; b.u32[2] = b0.z; b.u32[3] = b0.w;
                b.u32[4] = b1.x; b.u32[5] = b1.y; b.u32[6] = b1.z; b.u32[7] = b1.w;
                c = __builtin_amdgcn_wmma_f32_16x16x32_bf16(
                        false, a.v, false, b.v, (short)0, c, false, false);
            }
        }
    } else {
        for (int kd = 0; kd < 3; ++kd) {
            int jd = dd + kd - 1;                               // uniform
            if (jd < 0 || (jd & 1) || (jd >> 1) >= Di) continue;
            int di = (jd >> 1) + 1;
            for (int kh = 0; kh < 3; ++kh) {
                int jh = ho + kh - 1;                           // uniform
                if (jh < 0 || (jh & 1) || (jh >> 1) >= Hi) continue;
                int hi = (jh >> 1) + 1;
                for (int kw = 0; kw < 3; ++kw) {
                    int tap = kd * 9 + kh * 3 + kw;
                    int jw = wo + kw - 1;                       // per-lane
                    bool okw = (jw >= 0) && !(jw & 1) && ((jw >> 1) < Wi);
                    int wi = (jw >> 1) + 1;                     // stays in padded box
                    const unsigned short* ap =
                        (okw ? (in + (size_t)(di * planeC + hi * rowC + wi * Cin))
                             : zbuf) + khalf * 8;               // address cndmask
                    const unsigned short* wp = wbase + tap * Cin;
                    for (int c0 = 0; c0 < Cin; c0 += 32) {
                        uint4 a0 = *(const uint4*)(ap + c0);
                        uint4 a1 = *(const uint4*)(ap + c0 + 16);
                        uint4 b0 = *(const uint4*)(wp + c0);
                        uint4 b1 = *(const uint4*)(wp + c0 + 16);
                        FragBF a, b;
                        a.u32[0] = a0.x; a.u32[1] = a0.y; a.u32[2] = a0.z; a.u32[3] = a0.w;
                        a.u32[4] = a1.x; a.u32[5] = a1.y; a.u32[6] = a1.z; a.u32[7] = a1.w;
                        b.u32[0] = b0.x; b.u32[1] = b0.y; b.u32[2] = b0.z; b.u32[3] = b0.w;
                        b.u32[4] = b1.x; b.u32[5] = b1.y; b.u32[6] = b1.z; b.u32[7] = b1.w;
                        c = __builtin_amdgcn_wmma_f32_16x16x32_bf16(
                                false, a.v, false, b.v, (short)0, c, false, false);
                    }
                }
            }
        }
    }

    // D layout: VGPR r, lane l -> M(voxel) = r + (l>=16 ? 8 : 0), N(cout) = l&15
    int voxRow = (dd * Ho + ho) * Wo + tw * 16;
    if (outLayout == 0) {
#pragma unroll
        for (int r = 0; r < 8; ++r) {
            int mm = r + (khalf << 3);
            size_t oidx = (size_t)(voxRow + mm) * Cout + co;
            if (acc) out[oidx] += c[r];
            else     out[oidx]  = c[r];
        }
    } else if (co < Cout) {
        size_t V = (size_t)Do * Ho * Wo;
#pragma unroll
        for (int r = 0; r < 8; ++r) {
            int mm = r + (khalf << 3);
            size_t oidx = (size_t)co * V + voxRow + mm;
            if (acc) out[oidx] += c[r];
            else     out[oidx]  = c[r];
        }
    }
}

// ---------------------------------------------------------------------------
// BatchNorm statistics over channel-last unpadded raw f32: one block/channel
// ---------------------------------------------------------------------------
__global__ void __launch_bounds__(256) k_bn_stats(const float* __restrict__ x,
                                                  int C, int V,
                                                  float* __restrict__ stats) {
    int ch = blockIdx.x;
    float s = 0.f, s2 = 0.f;
    for (int v = threadIdx.x; v < V; v += 256) {
        float t = x[(size_t)v * C + ch];
        s += t; s2 += t * t;
    }
    __shared__ float sh1[256], sh2[256];
    sh1[threadIdx.x] = s; sh2[threadIdx.x] = s2;
    __syncthreads();
    for (int o = 128; o > 0; o >>= 1) {
        if (threadIdx.x < o) {
            sh1[threadIdx.x] += sh1[threadIdx.x + o];
            sh2[threadIdx.x] += sh2[threadIdx.x + o];
        }
        __syncthreads();
    }
    if (threadIdx.x == 0) {
        float m   = sh1[0] / (float)V;
        float var = sh2[0] / (float)V - m * m;
        stats[2 * ch]     = m;
        stats[2 * ch + 1] = rsqrtf(var + 1e-5f);
    }
}

// ---------------------------------------------------------------------------
// BN apply (+optional residual, +optional ReLU):
// unpadded channel-last f32 -> PADDED channel-last bf16 interior.
// Residual is read from the same padded geometry.
// ---------------------------------------------------------------------------
__global__ void __launch_bounds__(256) k_bn_apply(
    const float* __restrict__ x,
    const float* __restrict__ g, const float* __restrict__ b,
    const float* __restrict__ stats,
    const unsigned short* __restrict__ resid,
    unsigned short* __restrict__ outp,
    int lgC, int lgW, int lgH, int relu, int n)
{
    int i = blockIdx.x * 256 + threadIdx.x;
    if (i >= n) return;
    int C  = 1 << lgC;
    int ch = i & (C - 1);
    int vox = i >> lgC;
    int W = 1 << lgW, H = 1 << lgH;
    int w = vox & (W - 1);
    int h = (vox >> lgW) & (H - 1);
    int d = vox >> (lgW + lgH);
    int pidx = (((((d + 1) * (H + 2)) + (h + 1)) * (W + 2) + (w + 1)) << lgC) + ch;
    float v = (x[i] - stats[2 * ch]) * stats[2 * ch + 1] * g[ch] + b[ch];
    if (resid) v += bf2f(resid[pidx]);
    if (relu)  v = fmaxf(v, 0.f);
    outp[pidx] = f2bf(v);
}

// ---------------------------------------------------------------------------
// Host orchestration
// ---------------------------------------------------------------------------
extern "C" void kernel_launch(void* const* d_in, const int* in_sizes, int n_in,
                              void* d_out, int out_size, void* d_ws, size_t ws_size,
                              hipStream_t stream)
{
    (void)in_sizes; (void)n_in; (void)out_size; (void)ws_size;
    const float* conv_out = (const float*)d_in[0];

    // --- parameter leaf indices (setup_inputs dict order) ---
    int idx = 1;
    struct CB { int w, g, b; size_t wo; };
    auto take = [&](CB& cb) { cb.w = idx; cb.g = idx + 1; cb.b = idx + 2; cb.wo = 0; idx += 3; };
    CB dres0a, dres0b, dres1a, dres1b;
    take(dres0a); take(dres0b); take(dres1a); take(dres1b);
    CB hgc[3][6];
    for (int h = 0; h < 3; ++h)
        for (int j = 0; j < 6; ++j) take(hgc[h][j]);
    CB clscb[3]; int clswIdx[3];
    for (int c2 = 0; c2 < 3; ++c2) { take(clscb[c2]); clswIdx[c2] = idx++; }

    // --- geometry ---
    const int Df = 48, Hf = 64, Wf = 128; const int Vf = Df * Hf * Wf;     // 393216
    const int Dh = 24, Hh = 32, Wh = 64;  const int Vh = Dh * Hh * Wh;     // 49152
    const int Dq = 12, Hq = 16, Wq = 32;  const int Vq = Dq * Hq * Wq;     // 6144
    const size_t Pf = (size_t)(Df + 2) * (Hf + 2) * (Wf + 2);              // 429000
    const size_t Ph = (size_t)(Dh + 2) * (Hh + 2) * (Wh + 2);              // 58344
    const size_t Pq = (size_t)(Dq + 2) * (Hq + 2) * (Wq + 2);              // 8568

    // --- workspace layout ---
    char* wsb = (char*)d_ws;
    size_t off = 0;
    auto alloc = [&](size_t bytes) -> void* {
        off = (off + 255) & ~(size_t)255;
        void* p = wsb + off; off += bytes; return p;
    };

    unsigned short* wbf = (unsigned short*)alloc((size_t)5 * 1024 * 1024);

    // zero-initialized (halo) region: zbuf + all padded bf16 activations
    size_t zstart = (off + 255) & ~(size_t)255;
    unsigned short* zbuf  = (unsigned short*)alloc(256);
    unsigned short* cost  = (unsigned short*)alloc(Pf * 64 * 2);
    unsigned short* c0b   = (unsigned short*)alloc(Pf * 32 * 2);
    unsigned short* cost0 = (unsigned short*)alloc(Pf * 32 * 2);
    unsigned short* actA  = (unsigned short*)alloc(Pf * 32 * 2);
    unsigned short* tmpbf = (unsigned short*)alloc(Pf * 32 * 2);
    unsigned short* halfA = (unsigned short*)alloc(Ph * 64 * 2);
    unsigned short* pre1  = (unsigned short*)alloc(Ph * 64 * 2);
    unsigned short* preC  = (unsigned short*)alloc(Ph * 64 * 2);
    unsigned short* postP = (unsigned short*)alloc(Ph * 64 * 2);
    unsigned short* postC = (unsigned short*)alloc(Ph * 64 * 2);
    unsigned short* q1    = (unsigned short*)alloc(Pq * 64 * 2);
    unsigned short* q2    = (unsigned short*)alloc(Pq * 64 * 2);
    size_t zend = off;
    alloc(256);   // spacer so zero-fill rounding never touches the next buffer

    float* rawF  = (float*)alloc((size_t)32 * Vf * 4);
    float* rawH  = (float*)alloc((size_t)64 * Vh * 4);
    float* rawQ  = (float*)alloc((size_t)64 * Vq * 4);
    float* stats = (float*)alloc((size_t)2 * 64 * 4);

    // --- zero the activation region (halos must be 0; interiors overwritten) ---
    {
        unsigned int n16 = (unsigned int)((zend - zstart + 15) / 16);
        k_zero<<<(n16 + 255) / 256, 256, 0, stream>>>((uint4*)(wsb + zstart), n16);
    }

    // --- repack all conv weights to bf16 canonical [Coutp][27][Cin] ---
    size_t wo = 0;
    auto rpk = [&](int widx, int Cin, int Cout, int mode) -> size_t {
        int Coutp = ((Cout + 15) / 16) * 16;
        int n = Coutp * 27 * Cin;
        size_t o = wo;
        k_repack<<<(n + 255) / 256, 256, 0, stream>>>(
            (const float*)d_in[widx], wbf + o, Cin, Cout, mode, n);
        wo += (size_t)n;
        return o;
    };
    dres0a.wo = rpk(dres0a.w, 64, 32, 0);
    dres0b.wo = rpk(dres0b.w, 32, 32, 0);
    dres1a.wo = rpk(dres1a.w, 32, 32, 0);
    dres1b.wo = rpk(dres1b.w, 32, 32, 0);
    for (int h = 0; h < 3; ++h) {
        hgc[h][0].wo = rpk(hgc[h][0].w, 32, 64, 0);   // conv1
        hgc[h][1].wo = rpk(hgc[h][1].w, 64, 64, 0);   // conv2
        hgc[h][2].wo = rpk(hgc[h][2].w, 64, 64, 0);   // conv3
        hgc[h][3].wo = rpk(hgc[h][3].w, 64, 64, 0);   // conv4
        hgc[h][4].wo = rpk(hgc[h][4].w, 64, 64, 1);   // conv5 (T)
        hgc[h][5].wo = rpk(hgc[h][5].w, 64, 32, 1);   // conv6 (T)
    }
    size_t clswo[3];
    for (int c2 = 0; c2 < 3; ++c2) {
        clscb[c2].wo = rpk(clscb[c2].w, 32, 32, 0);
        clswo[c2]    = rpk(clswIdx[c2], 32, 19, 0);   // padded to 32
    }

    // log2 helpers per resolution
    auto lg = [](int v) { int l = 0; while ((1 << l) < v) ++l; return l; };
    const int lgWf = lg(Wf), lgHf = lg(Hf);
    const int lgWh = lg(Wh), lgHh = lg(Hh);
    const int lgWq = lg(Wq), lgHq = lg(Hq);

    // --- conv + BN(+resid,+relu) helper ---
    auto convbn = [&](const unsigned short* inb, CB& p,
                      int Cin, int Hpi, int Wpi,          // padded input dims
                      int Di, int Hi, int Wi,             // unpadded input dims
                      int Cout, int Do, int Ho, int Wo,
                      int lgWo, int lgHo,
                      int stride, int mode, float* raw,
                      const unsigned short* resid, unsigned short* outb, int relu) {
        int nT = ((Cout + 15) / 16) * Do * Ho * (Wo / 16);
        k_conv<<<(nT + 7) / 8, 256, 0, stream>>>(inb, wbf + p.wo, zbuf, raw,
                                                 Cin, Hpi, Wpi, Di, Hi, Wi,
                                                 Cout, Do, Ho, Wo,
                                                 stride, mode, 0, 0, nT);
        int V = Do * Ho * Wo;
        k_bn_stats<<<Cout, 256, 0, stream>>>(raw, Cout, V, stats);
        int n = Cout * V;
        int lgC = (Cout == 64) ? 6 : 5;
        k_bn_apply<<<(n + 255) / 256, 256, 0, stream>>>(
            raw, (const float*)d_in[p.g], (const float*)d_in[p.b],
            stats, resid, outb, lgC, lgWo, lgHo, relu, n);
    };

    // --- cost volume (padded interior) ---
    {
        int n = 64 * Vf;
        k_cost<<<(n + 255) / 256, 256, 0, stream>>>(conv_out, cost, n);
    }

    // --- dres0 / dres1 ---
    convbn(cost,  dres0a, 64, Hf+2, Wf+2, Df, Hf, Wf, 32, Df, Hf, Wf, lgWf, lgHf, 1, 0, rawF, nullptr, tmpbf, 1);
    convbn(tmpbf, dres0b, 32, Hf+2, Wf+2, Df, Hf, Wf, 32, Df, Hf, Wf, lgWf, lgHf, 1, 0, rawF, nullptr, c0b,   1);
    convbn(c0b,   dres1a, 32, Hf+2, Wf+2, Df, Hf, Wf, 32, Df, Hf, Wf, lgWf, lgHf, 1, 0, rawF, nullptr, tmpbf, 1);
    convbn(tmpbf, dres1b, 32, Hf+2, Wf+2, Df, Hf, Wf, 32, Df, Hf, Wf, lgWf, lgHf, 1, 0, rawF, c0b,     cost0, 0);

    // --- hourglass ---
    auto hourglass = [&](const unsigned short* x, CB* P,
                         const unsigned short* presqu, const unsigned short* postsqu,
                         unsigned short* preOut, unsigned short* postOut,
                         unsigned short* outOut) {
        convbn(x,      P[0], 32, Hf+2, Wf+2, Df, Hf, Wf, 64, Dh, Hh, Wh, lgWh, lgHh, 2, 0, rawH, nullptr, halfA, 1);
        convbn(halfA,  P[1], 64, Hh+2, Wh+2, Dh, Hh, Wh, 64, Dh, Hh, Wh, lgWh, lgHh, 1, 0, rawH, postsqu, preOut, 1);
        convbn(preOut, P[2], 64, Hh+2, Wh+2, Dh, Hh, Wh, 64, Dq, Hq, Wq, lgWq, lgHq, 2, 0, rawQ, nullptr, q1,     1);
        convbn(q1,     P[3], 64, Hq+2, Wq+2, Dq, Hq, Wq, 64, Dq, Hq, Wq, lgWq, lgHq, 1, 0, rawQ, nullptr, q2,     1);
        const unsigned short* addsrc = presqu ? presqu : preOut;
        convbn(q2,     P[4], 64, Hq+2, Wq+2, Dq, Hq, Wq, 64, Dh, Hh, Wh, lgWh, lgHh, 1, 1, rawH, addsrc,  postOut, 1);
        convbn(postOut,P[5], 64, Hh+2, Wh+2, Dh, Hh, Wh, 32, Df, Hf, Wf, lgWf, lgHf, 1, 1, rawF, cost0,   outOut, 0);
    };

    // --- classifier head: raw conv output accumulated directly into d_out ---
    auto clshead = [&](const unsigned short* x, CB& cb, size_t wofs, int accflag) {
        convbn(x, cb, 32, Hf+2, Wf+2, Df, Hf, Wf, 32, Df, Hf, Wf, lgWf, lgHf, 1, 0, rawF, nullptr, tmpbf, 1);
        int nT = 2 * Df * Hf * (Wf / 16);   // Coutp=32 -> 2 cout tiles
        k_conv<<<(nT + 7) / 8, 256, 0, stream>>>(tmpbf, wbf + wofs, zbuf, (float*)d_out,
                                                 32, Hf+2, Wf+2, Df, Hf, Wf,
                                                 19, Df, Hf, Wf,
                                                 1, 0, 1, accflag, nT);
    };

    hourglass(cost0, hgc[0], nullptr, nullptr, pre1, postP, actA);   // out1
    clshead(actA, clscb[0], clswo[0], 0);
    hourglass(actA, hgc[1], pre1, postP, preC, postC, actA);         // out2
    clshead(actA, clscb[1], clswo[1], 1);
    hourglass(actA, hgc[2], pre1, postC, preC, postP, actA);         // out3
    clshead(actA, clscb[2], clswo[2], 1);
}
